// SpanScorer_1975684956404
// MI455X (gfx1250) — matrix operand order repaired and checked by reading, not compiled, not run on hardware
//
#include <hip/hip_runtime.h>
#include <hip/hip_fp16.h>

typedef __attribute__((ext_vector_type(16))) _Float16 v16h;
typedef __attribute__((ext_vector_type(8)))  float    v8f;

#define NPAD 160   // padded hidden width (150 -> 160), 10 WMMA N-tiles
#define NT   10
#define HIDR 150

union FragH { v16h h; unsigned u[8]; };
union H2U { _Float16 f; unsigned short s; };

// ---- f32 -> f16 with row/col zero padding -------------------------------
__global__ void cvt_pad_f16(const float* __restrict__ src, _Float16* __restrict__ dst,
                            int M, int Ksrc, int Kpad, int Mpad) {
  long i = (long)blockIdx.x * blockDim.x + threadIdx.x;
  long total = (long)Mpad * Kpad;
  if (i >= total) return;
  int r = (int)(i / Kpad), c = (int)(i % Kpad);
  float v = (r < M && c < Ksrc) ? src[(long)r * Ksrc + c] : 0.f;
  dst[i] = (_Float16)v;
}

// ---- pack weight block [K x Nsrc] (f32, row-major, Nsrc-col stride) into
// ---- FRAGMENT-MAJOR WMMA operands:
// ---- PB dword index = ((kb*NT + t)*32 + lane)*8 + v
// ---- lane: h = lane>>4 selects K half (0-15 / 16-31), n = t*16 + (lane&15)
// ---- v:    K pair = kb*32 + h*16 + v*2  -> dword = (B[k][n], B[k+1][n])
__global__ void pack_b_frag(const float* __restrict__ W, int rowOff, int K, int Nsrc,
                            int Kpad, unsigned* __restrict__ PB) {
  int i = blockIdx.x * blockDim.x + threadIdx.x;
  int total = (Kpad / 32) * NT * 256;          // dwords
  if (i >= total) return;
  int v    = i & 7;
  int lane = (i >> 3) & 31;
  int blk  = i >> 8;
  int t    = blk % NT;
  int kb   = blk / NT;
  int h    = lane >> 4;
  int n    = t * 16 + (lane & 15);
  int k    = kb * 32 + h * 16 + v * 2;
  float lo = (k     < K && n < Nsrc) ? W[(long)(rowOff + k    ) * Nsrc + n] : 0.f;
  float hi = (k + 1 < K && n < Nsrc) ? W[(long)(rowOff + k + 1) * Nsrc + n] : 0.f;
  H2U l, hu; l.f = (_Float16)lo; hu.f = (_Float16)hi;
  PB[i] = ((unsigned)hu.s << 16) | (unsigned)l.s;
}

// ---- generic WMMA GEMM: C[M x 160] = act(A[M x K]f16 @ B + bias) --------
// one wave = 16-row strip x all 10 N-tiles; M must be multiple of 128
__global__ void gemm_wmma_f16(const _Float16* __restrict__ A, int lda, int K,
                              const unsigned* __restrict__ PB,
                              const float* __restrict__ bias, int nBias,
                              void* __restrict__ Cout, int ldc,
                              int storeHalf, int applyRelu) {
  const int lane = threadIdx.x & 31;
  const int wave = threadIdx.x >> 5;
  const int m0   = (blockIdx.x * 8 + wave) * 16;
  const int ml   = lane & 15;      // row within tile (A) / col within tile (B,C)
  const int hh   = lane >> 4;      // lane-half selector
  const int m    = m0 + ml;

  v8f acc[NT];
  v8f zero = {0.f,0.f,0.f,0.f,0.f,0.f,0.f,0.f};
#pragma unroll
  for (int t = 0; t < NT; t++) acc[t] = zero;

  const unsigned* __restrict__ pbLane = PB + (size_t)lane * 8;

  int kb = 0;
  for (int kk = 0; kk < K; kk += 32, kb++) {
    FragH a;
#pragma unroll
    for (int v = 0; v < 8; v++) {           // A: contiguous k = 16*hh + 2v, same as B pack
      int k = kk + hh * 16 + v * 2;
      a.u[v] = *(const unsigned*)(A + (size_t)m * lda + k);
    }
    // issue ALL B-tile fragment loads first (2x b128 each), then the WMMA
    // chain -> loads overlap with matrix math instead of full drains
    FragH b[NT];
    const unsigned* __restrict__ pbk = pbLane + (size_t)kb * NT * 256;
#pragma unroll
    for (int t = 0; t < NT; t++) {
#pragma unroll
      for (int v = 0; v < 8; v++)
        b[t].u[v] = pbk[t * 256 + v];
    }
#pragma unroll
    for (int t = 0; t < NT; t++) {
      acc[t] = __builtin_amdgcn_wmma_f32_16x16x32_f16(
          false, a.h, false, b[t].h, (short)0, acc[t], false, false);
    }
  }

#pragma unroll
  for (int t = 0; t < NT; t++) {
    int col = t * 16 + ml;
    float bv = (bias && col < nBias) ? bias[col] : 0.f;
#pragma unroll
    for (int r = 0; r < 8; r++) {           // C/D layout: rows r and r+8 by lane half
      int row = m0 + r + 8 * hh;
      float c = acc[t][r] + bv;
      if (applyRelu) c = fmaxf(c, 0.f);
      if (storeHalf) ((_Float16*)Cout)[(size_t)row * ldc + col] = (_Float16)c;
      else           ((float*)Cout)[(size_t)row * ldc + col] = c;
    }
  }
}

// ---- attn layer 3: scores[i] = A2[i,:150] . aW3 + ab3 -------------------
__global__ void attn_out(const _Float16* __restrict__ A2, const float* __restrict__ aW3,
                         const float* __restrict__ ab3, float* __restrict__ scores, int T) {
  int i = blockIdx.x * blockDim.x + threadIdx.x;
  if (i >= T) return;
  const _Float16* row = A2 + (size_t)i * NPAD;
  float s = ab3[0];
  for (int j = 0; j < HIDR; j++) s += (float)row[j] * aW3[j];
  scores[i] = s;
}

__device__ __forceinline__ int dtoi_dev(int d) {
  const int bins[8] = {1, 2, 3, 4, 8, 16, 32, 64};
  int c = 0;
  for (int i = 0; i < 8; i++) c += (d >= bins[i]) ? 1 : 0;
  return c;
}

// ---- wbias[n][j] = sb1[j] + width_table[dtoi(n)] . sW1[1100:1120, j] ----
__global__ void width_bias(const float* __restrict__ wt, const float* __restrict__ sW1,
                           const float* __restrict__ sb1, float* __restrict__ wb) {
  int i = blockIdx.x * blockDim.x + threadIdx.x;
  if (i >= 10 * NPAD) return;
  int n = i / NPAD + 1, j = i % NPAD;
  float v = 0.f;
  if (j < HIDR) {
    v = sb1[j];
    int r = dtoi_dev(n);
    for (int d = 0; d < 20; d++)
      v += wt[r * 20 + d] * sW1[(size_t)(1100 + d) * HIDR + j];
  }
  wb[i] = v;
}

// ---- fused span stage: softmax + gather-combine -> H1 (relu, f16) -------
__global__ void span_h1(const float* __restrict__ scores,
                        const float* __restrict__ P1, const float* __restrict__ P2,
                        const float* __restrict__ E,  const float* __restrict__ wb,
                        _Float16* __restrict__ H1, int T, int S, int SPAD) {
  int lane = threadIdx.x & 31;
  int idx  = blockIdx.x * 8 + (threadIdx.x >> 5);
  if (idx >= SPAD) return;
  if (idx >= S) {  // zero pad rows so downstream GEMM needs no guards
#pragma unroll
    for (int jj = 0; jj < NPAD; jj += 32)
      H1[(size_t)idx * NPAD + jj + lane] = (_Float16)0.f;
    return;
  }
  int n = 1, off = 0;
  for (; n <= 10; n++) { int w = T - n + 1; if (idx < off + w) break; off += w; }
  int s = idx - off, e = s + n - 1;

  float a[10];
  float mx = -3.4e38f;
  for (int k = 0; k < n; k++) { a[k] = scores[s + k]; mx = fmaxf(mx, a[k]); }
  float sum = 0.f;
  for (int k = 0; k < n; k++) { a[k] = __expf(a[k] - mx); sum += a[k]; }
  float inv = 1.f / sum;
  for (int k = 0; k < n; k++) a[k] *= inv;

#pragma unroll
  for (int jj = 0; jj < NPAD; jj += 32) {
    int j = jj + lane;
    float acc = P1[(size_t)s * NPAD + j] + P2[(size_t)e * NPAD + j]
              + wb[(n - 1) * NPAD + j];
    for (int k = 0; k < n; k++) acc += a[k] * E[(size_t)(s + k) * NPAD + j];
    acc = fmaxf(acc, 0.f);
    H1[(size_t)idx * NPAD + j] = (_Float16)acc;
  }
}

// ---- sm layer 3 ---------------------------------------------------------
__global__ void final_out(const _Float16* __restrict__ H2, const float* __restrict__ sW3,
                          const float* __restrict__ sb3, float* __restrict__ out, int S) {
  int i = blockIdx.x * blockDim.x + threadIdx.x;
  if (i >= S) return;
  const _Float16* row = H2 + (size_t)i * NPAD;
  float v = sb3[0];
  for (int j = 0; j < HIDR; j++) v += (float)row[j] * sW3[j];
  out[i] = v;
}

extern "C" void kernel_launch(void* const* d_in, const int* in_sizes, int n_in,
                              void* d_out, int out_size, void* d_ws, size_t ws_size,
                              hipStream_t stream) {
  const float* embeds = (const float*)d_in[0];
  const float* states = (const float*)d_in[1];
  const float* aW1 = (const float*)d_in[2];
  const float* ab1 = (const float*)d_in[3];
  const float* aW2 = (const float*)d_in[4];
  const float* ab2 = (const float*)d_in[5];
  const float* aW3 = (const float*)d_in[6];
  const float* ab3 = (const float*)d_in[7];
  const float* wtab = (const float*)d_in[8];
  const float* sW1 = (const float*)d_in[9];
  const float* sb1 = (const float*)d_in[10];
  const float* sW2 = (const float*)d_in[11];
  const float* sb2 = (const float*)d_in[12];
  const float* sW3 = (const float*)d_in[13];
  const float* sb3 = (const float*)d_in[14];

  const int T    = in_sizes[1] / 400;
  const int KS = 416, KE = 320, KH = 160;      // padded K dims (mult of 32)
  const int Tpad = ((T + 127) / 128) * 128;
  const int S    = 10 * T - 45;
  const int SPAD = ((S + 127) / 128) * 128;

  char* p = (char*)d_ws;
  auto alloc = [&](size_t bytes) -> char* {
    char* r = p; p += (bytes + 255) & ~(size_t)255; return r;
  };
  // fragment-major B size in bytes: (Kpad/32) * NT * 256 dwords * 4
  auto pbBytes = [](int Kpad) -> size_t { return (size_t)(Kpad / 32) * NT * 256 * 4; };

  _Float16* states_h = (_Float16*)alloc((size_t)Tpad * KS * 2);
  _Float16* embeds_h = (_Float16*)alloc((size_t)Tpad * KE * 2);
  unsigned* BPa1  = (unsigned*)alloc(pbBytes(KS));
  unsigned* BPa2  = (unsigned*)alloc(pbBytes(KH));
  unsigned* BPs1a = (unsigned*)alloc(pbBytes(KS));
  unsigned* BPs1b = (unsigned*)alloc(pbBytes(KS));
  unsigned* BPs1c = (unsigned*)alloc(pbBytes(KE));
  unsigned* BPs2  = (unsigned*)alloc(pbBytes(KH));
  float* P1 = (float*)alloc((size_t)Tpad * NPAD * 4);
  float* P2 = (float*)alloc((size_t)Tpad * NPAD * 4);
  float* Ep = (float*)alloc((size_t)Tpad * NPAD * 4);
  _Float16* A1h = (_Float16*)alloc((size_t)Tpad * NPAD * 2);
  _Float16* A2h = (_Float16*)alloc((size_t)Tpad * NPAD * 2);
  float* scores = (float*)alloc((size_t)Tpad * 4);
  float* wb = (float*)alloc(10 * NPAD * 4);
  _Float16* H1 = (_Float16*)alloc((size_t)SPAD * NPAD * 2);
  _Float16* H2 = (_Float16*)alloc((size_t)SPAD * NPAD * 2);

  { long tot = (long)Tpad * KS;
    cvt_pad_f16<<<(int)((tot + 255) / 256), 256, 0, stream>>>(states, states_h, T, 400, KS, Tpad); }
  { long tot = (long)Tpad * KE;
    cvt_pad_f16<<<(int)((tot + 255) / 256), 256, 0, stream>>>(embeds, embeds_h, T, 300, KE, Tpad); }

  { int tot = (KS / 32) * NT * 256; pack_b_frag<<<(tot + 255) / 256, 256, 0, stream>>>(aW1,   0, 400, 150, KS, BPa1); }
  { int tot = (KH / 32) * NT * 256; pack_b_frag<<<(tot + 255) / 256, 256, 0, stream>>>(aW2,   0, 150, 150, KH, BPa2); }
  { int tot = (KS / 32) * NT * 256; pack_b_frag<<<(tot + 255) / 256, 256, 0, stream>>>(sW1,   0, 400, 150, KS, BPs1a); }
  { int tot = (KS / 32) * NT * 256; pack_b_frag<<<(tot + 255) / 256, 256, 0, stream>>>(sW1, 400, 400, 150, KS, BPs1b); }
  { int tot = (KE / 32) * NT * 256; pack_b_frag<<<(tot + 255) / 256, 256, 0, stream>>>(sW1, 800, 300, 150, KE, BPs1c); }
  { int tot = (KH / 32) * NT * 256; pack_b_frag<<<(tot + 255) / 256, 256, 0, stream>>>(sW2,   0, 150, 150, KH, BPs2); }
  width_bias<<<(10 * NPAD + 255) / 256, 256, 0, stream>>>(wtab, sW1, sb1, wb);

  const int mb = Tpad / 128;
  gemm_wmma_f16<<<mb, 256, 0, stream>>>(states_h, KS, KS, BPa1,  ab1, 150, A1h, NPAD, 1, 1);
  gemm_wmma_f16<<<mb, 256, 0, stream>>>(states_h, KS, KS, BPs1a, nullptr, 0, P1, NPAD, 0, 0);
  gemm_wmma_f16<<<mb, 256, 0, stream>>>(states_h, KS, KS, BPs1b, nullptr, 0, P2, NPAD, 0, 0);
  gemm_wmma_f16<<<mb, 256, 0, stream>>>(embeds_h, KE, KE, BPs1c, nullptr, 0, Ep, NPAD, 0, 0);
  gemm_wmma_f16<<<mb, 256, 0, stream>>>(A1h, NPAD, KH, BPa2, ab2, 150, A2h, NPAD, 1, 1);
  attn_out<<<(T + 255) / 256, 256, 0, stream>>>(A2h, aW3, ab3, scores, T);

  span_h1<<<SPAD / 8, 256, 0, stream>>>(scores, P1, P2, Ep, wb, H1, T, S, SPAD);

  gemm_wmma_f16<<<SPAD / 128, 256, 0, stream>>>(H1, NPAD, KH, BPs2, sb2, 150, H2, NPAD, 1, 1);
  final_out<<<(S + 255) / 256, 256, 0, stream>>>(H2, sW3, sb3, (float*)d_out, S);
}
